// EMPSNLayer_9302899163194
// MI455X (gfx1250) — compile-verified, matrix operand order checked by reading, not run on hardware
//
#include <hip/hip_runtime.h>
#include <stdint.h>

typedef __attribute__((ext_vector_type(16))) __bf16 bf16x16;
typedef __attribute__((ext_vector_type(8)))  float  floatx8;

#define CCH 128              // channels
#define EKP 288              // edge-MLP fin padded to multiple of 32 (259/262 -> 288)

// fast elementwise epilogues: v_exp_f32 + v_rcp_f32 instead of IEEE div expansion
__device__ __forceinline__ float sigm_f(float x) {
    return __builtin_amdgcn_rcpf(1.f + __expf(-x));
}
__device__ __forceinline__ float silu_f(float x) {
    return x * __builtin_amdgcn_rcpf(1.f + __expf(-x));
}

// CDNA5 16-bit A/B fragment layout (ISA 7.12.2): lane L holds row/col (L&15),
// K elements: i<8 -> k = half*8 + i ; i>=8 -> k = 16 + half*8 + (i-8).
__device__ __forceinline__ bf16x16 load_frag(const __bf16* __restrict__ base,
                                             int stride, int rc, int k0, int half) {
    bf16x16 v;
#pragma unroll
    for (int i = 0; i < 16; ++i) {
        int kl = (i < 8) ? (half * 8 + i) : (8 + half * 8 + i);
        v[i] = base[(size_t)rc * stride + k0 + kl];
    }
    return v;
}

// ---------------- workspace prep kernels ----------------

__global__ void zero_kernel(float* __restrict__ p, size_t n) {
    size_t i = (size_t)blockIdx.x * blockDim.x + threadIdx.x;
    size_t stride = (size_t)gridDim.x * blockDim.x;
    for (; i < n; i += stride) p[i] = 0.f;
}

// src: f32 [K][Ncols] row-major  ->  dst: bf16 [Ncols][Kpad] (transposed, zero-padded)
__global__ void convert_wt(const float* __restrict__ src, __bf16* __restrict__ dst,
                           int K, int Kpad, int Ncols) {
    size_t idx = (size_t)blockIdx.x * blockDim.x + threadIdx.x;
    size_t total = (size_t)Ncols * Kpad;
    if (idx >= total) return;
    int n = (int)(idx / Kpad);
    int k = (int)(idx % Kpad);
    float v = (k < K) ? src[(size_t)k * Ncols + n] : 0.f;
    dst[idx] = (__bf16)v;
}

// ---------------- edge MLP + gate + scatter-add ----------------
// 128 threads = 4 waves; each wave owns one 16-edge tile.
__global__ __launch_bounds__(128) void edge_mlp_kernel(
    const float* __restrict__ xsrc, const float* __restrict__ xtgt,
    const int* __restrict__ send, const int* __restrict__ recv,
    const float* __restrict__ inv, int ninv,
    const __bf16* __restrict__ w1t,          // bf16 [128][EKP]
    const float* __restrict__ b1,
    const __bf16* __restrict__ w2t,          // bf16 [128][128]
    const float* __restrict__ b2,
    const float* __restrict__ wg, const float* __restrict__ bg,
    float* __restrict__ agg, int E)
{
    __shared__ __bf16 sh[4 * (16 * EKP + 16 * CCH)];
    const int lane = threadIdx.x & 31;
    const int wave = threadIdx.x >> 5;
    const int half = lane >> 4;
    const int c    = lane & 15;
    __bf16* arow = sh + wave * (16 * EKP + 16 * CCH);
    __bf16* mbuf = arow + 16 * EKP;

    const long tile  = (long)blockIdx.x * 4 + wave;
    const long ebase = tile * 16;

    // ---- gather 16 rows: [x_src | x_tgt | inv | pad] -> bf16 LDS ----
    for (int m = 0; m < 16; ++m) {
        long e = ebase + m;
        bool valid = (e < E);
        int s = valid ? send[e] : 0;
        float4 vs = make_float4(0.f, 0.f, 0.f, 0.f);
        float4 vt = make_float4(0.f, 0.f, 0.f, 0.f);
        if (valid) {
            int r = recv[e];
            vs = ((const float4*)(xsrc + (size_t)s * CCH))[lane];
            vt = ((const float4*)(xtgt + (size_t)r * CCH))[lane];
        }
        __bf16* row = arow + m * EKP;
        row[lane * 4 + 0]       = (__bf16)vs.x;
        row[lane * 4 + 1]       = (__bf16)vs.y;
        row[lane * 4 + 2]       = (__bf16)vs.z;
        row[lane * 4 + 3]       = (__bf16)vs.w;
        row[CCH + lane * 4 + 0] = (__bf16)vt.x;
        row[CCH + lane * 4 + 1] = (__bf16)vt.y;
        row[CCH + lane * 4 + 2] = (__bf16)vt.z;
        row[CCH + lane * 4 + 3] = (__bf16)vt.w;
        float iv = (valid && lane < ninv) ? inv[(size_t)e * ninv + lane] : 0.f;
        row[2 * CCH + lane] = (__bf16)iv;                 // covers 256..287 incl. pad
    }
    __syncthreads();

    // ---- layer 1: m1 = silu(A @ W1 + b1), staged to LDS as bf16 ----
    for (int t = 0; t < 8; ++t) {
        float bias = b1[t * 16 + c];
        floatx8 acc;
#pragma unroll
        for (int r = 0; r < 8; ++r) acc[r] = bias;
        for (int k0 = 0; k0 < EKP; k0 += 32) {
            bf16x16 a = load_frag(arow, EKP, c, k0, half);
            bf16x16 b = load_frag(w1t,  EKP, t * 16 + c, k0, half);
            acc = __builtin_amdgcn_wmma_f32_16x16x32_bf16(false, a, false, b,
                                                          (short)0, acc, false, false);
        }
#pragma unroll
        for (int r = 0; r < 8; ++r)
            mbuf[(half * 8 + r) * CCH + (t * 16 + c)] = (__bf16)silu_f(acc[r]);
    }
    __syncthreads();

    // ---- layer 2: m2 = silu(m1 @ W2 + b2), keep all 8 N-tiles live ----
    floatx8 acc2[8];
    for (int t = 0; t < 8; ++t) {
        float bias = b2[t * 16 + c];
#pragma unroll
        for (int r = 0; r < 8; ++r) acc2[t][r] = bias;
        for (int k0 = 0; k0 < CCH; k0 += 32) {
            bf16x16 a = load_frag(mbuf, CCH, c, k0, half);
            bf16x16 b = load_frag(w2t,  CCH, t * 16 + c, k0, half);
            acc2[t] = __builtin_amdgcn_wmma_f32_16x16x32_bf16(false, a, false, b,
                                                              (short)0, acc2[t], false, false);
        }
#pragma unroll
        for (int r = 0; r < 8; ++r) acc2[t][r] = silu_f(acc2[t][r]);
    }

    // ---- gate = sigmoid(m2 @ Wg + bg); rows live across 16-lane halves ----
    float wgv[8];
#pragma unroll
    for (int t = 0; t < 8; ++t) wgv[t] = wg[t * 16 + c];
    float bgs = bg[0];
    float gate[8];
#pragma unroll
    for (int r = 0; r < 8; ++r) {
        float part = 0.f;
#pragma unroll
        for (int t = 0; t < 8; ++t) part += acc2[t][r] * wgv[t];
        part += __shfl_xor(part, 1, 32);
        part += __shfl_xor(part, 2, 32);
        part += __shfl_xor(part, 4, 32);
        part += __shfl_xor(part, 8, 32);
        gate[r] = sigm_f(part + bgs);
    }

    // ---- scatter-add gated messages into aggregate ----
#pragma unroll
    for (int r = 0; r < 8; ++r) {
        int  M = half * 8 + r;
        long e = ebase + M;
        if (e < E) {
            int ri = recv[e];
            float* dst = agg + (size_t)ri * CCH + c;
#pragma unroll
            for (int t = 0; t < 8; ++t)
                atomicAdd(dst + t * 16, acc2[t][r] * gate[r]);
        }
    }
}

// ---------------- cell update MLP: out = f + (silu([f|silu(g1)|silu(g2)] @ W1 + b1) @ W2 + b2)
// 64 threads = 2 waves; each wave owns one 16-cell tile.
__global__ __launch_bounds__(64) void update_kernel(
    const float* __restrict__ f,
    const float* __restrict__ g1, const float* __restrict__ g2,   // g2 may be null
    const __bf16* __restrict__ w1t, int kpad,                      // 256 or 384
    const float* __restrict__ b1,
    const __bf16* __restrict__ w2t, const float* __restrict__ b2,
    float* __restrict__ out, int n)
{
    __shared__ __bf16 sh[2 * (16 * 384 + 16 * CCH)];
    const int lane = threadIdx.x & 31;
    const int wave = threadIdx.x >> 5;
    const int half = lane >> 4;
    const int c    = lane & 15;
    __bf16* arow = sh + wave * (16 * 384 + 16 * CCH);
    __bf16* mbuf = arow + 16 * 384;

    const long tile  = (long)blockIdx.x * 2 + wave;
    const long ibase = tile * 16;

    for (int m = 0; m < 16; ++m) {
        long i = ibase + m;
        bool valid = (i < n);
        float4 vf = make_float4(0.f, 0.f, 0.f, 0.f);
        float4 v1 = make_float4(0.f, 0.f, 0.f, 0.f);
        float4 v2 = make_float4(0.f, 0.f, 0.f, 0.f);
        if (valid) {
            vf = ((const float4*)(f  + (size_t)i * CCH))[lane];
            v1 = ((const float4*)(g1 + (size_t)i * CCH))[lane];
            if (g2) v2 = ((const float4*)(g2 + (size_t)i * CCH))[lane];
        }
        __bf16* row = arow + m * kpad;
        row[lane * 4 + 0] = (__bf16)vf.x;  row[lane * 4 + 1] = (__bf16)vf.y;
        row[lane * 4 + 2] = (__bf16)vf.z;  row[lane * 4 + 3] = (__bf16)vf.w;
        row[CCH + lane * 4 + 0] = (__bf16)silu_f(v1.x);
        row[CCH + lane * 4 + 1] = (__bf16)silu_f(v1.y);
        row[CCH + lane * 4 + 2] = (__bf16)silu_f(v1.z);
        row[CCH + lane * 4 + 3] = (__bf16)silu_f(v1.w);
        if (g2) {
            row[2 * CCH + lane * 4 + 0] = (__bf16)silu_f(v2.x);
            row[2 * CCH + lane * 4 + 1] = (__bf16)silu_f(v2.y);
            row[2 * CCH + lane * 4 + 2] = (__bf16)silu_f(v2.z);
            row[2 * CCH + lane * 4 + 3] = (__bf16)silu_f(v2.w);
        }
    }
    __syncthreads();

    for (int t = 0; t < 8; ++t) {
        float bias = b1[t * 16 + c];
        floatx8 acc;
#pragma unroll
        for (int r = 0; r < 8; ++r) acc[r] = bias;
        for (int k0 = 0; k0 < kpad; k0 += 32) {
            bf16x16 a = load_frag(arow, kpad, c, k0, half);
            bf16x16 b = load_frag(w1t,  kpad, t * 16 + c, k0, half);
            acc = __builtin_amdgcn_wmma_f32_16x16x32_bf16(false, a, false, b,
                                                          (short)0, acc, false, false);
        }
#pragma unroll
        for (int r = 0; r < 8; ++r)
            mbuf[(half * 8 + r) * CCH + (t * 16 + c)] = (__bf16)silu_f(acc[r]);
    }
    __syncthreads();

    for (int t = 0; t < 8; ++t) {
        float bias = b2[t * 16 + c];
        floatx8 acc;
#pragma unroll
        for (int r = 0; r < 8; ++r) acc[r] = bias;
        for (int k0 = 0; k0 < CCH; k0 += 32) {
            bf16x16 a = load_frag(mbuf, CCH, c, k0, half);
            bf16x16 b = load_frag(w2t,  CCH, t * 16 + c, k0, half);
            acc = __builtin_amdgcn_wmma_f32_16x16x32_bf16(false, a, false, b,
                                                          (short)0, acc, false, false);
        }
#pragma unroll
        for (int r = 0; r < 8; ++r) {
            int  M = half * 8 + r;
            long i = ibase + M;
            if (i < n) {
                size_t o = (size_t)i * CCH + t * 16 + c;
                out[o] = f[o] + acc[r];       // residual, no activation
            }
        }
    }
}

// ---------------- host launcher ----------------

extern "C" void kernel_launch(void* const* d_in, const int* in_sizes, int n_in,
                              void* d_out, int out_size, void* d_ws, size_t ws_size,
                              hipStream_t stream) {
    (void)n_in; (void)out_size; (void)ws_size;
    const float* f0 = (const float*)d_in[0];
    const float* f1 = (const float*)d_in[1];
    const float* f2 = (const float*)d_in[2];
    const int N = in_sizes[0] / CCH;
    const size_t NC = (size_t)N * CCH;

    // workspace: 4 f32 aggregation buffers, then bf16 transposed weights
    float* agg = (float*)d_ws;
    float* aggp[4] = { agg, agg + NC, agg + 2 * NC, agg + 3 * NC };
    __bf16* wcur = (__bf16*)((char*)d_ws + 4 * NC * sizeof(float));

    zero_kernel<<<4096, 256, 0, stream>>>(agg, 4 * NC);

    const int relbase[4] = { 3, 12, 21, 30 };
    int E[4], ninv[4];
    __bf16 *w1t[4], *w2t[4];
    for (int r = 0; r < 4; ++r) {
        int b = relbase[r];
        E[r] = in_sizes[b];
        ninv[r] = in_sizes[b + 2] / E[r];
        int K1 = 2 * CCH + ninv[r];
        w1t[r] = wcur; wcur += (size_t)CCH * EKP;
        convert_wt<<<(CCH * EKP + 255) / 256, 256, 0, stream>>>(
            (const float*)d_in[b + 3], w1t[r], K1, EKP, CCH);
        w2t[r] = wcur; wcur += (size_t)CCH * CCH;
        convert_wt<<<(CCH * CCH + 255) / 256, 256, 0, stream>>>(
            (const float*)d_in[b + 5], w2t[r], CCH, CCH, CCH);
    }

    const int ubase[3] = { 39, 43, 47 };
    const int ukpad[3] = { 256, 384, 256 };
    __bf16 *uw1t[3], *uw2t[3];
    for (int u = 0; u < 3; ++u) {
        int b = ubase[u];
        uw1t[u] = wcur; wcur += (size_t)CCH * ukpad[u];
        convert_wt<<<(CCH * ukpad[u] + 255) / 256, 256, 0, stream>>>(
            (const float*)d_in[b], uw1t[u], ukpad[u], ukpad[u], CCH);
        uw2t[u] = wcur; wcur += (size_t)CCH * CCH;
        convert_wt<<<(CCH * CCH + 255) / 256, 256, 0, stream>>>(
            (const float*)d_in[b + 2], uw2t[u], CCH, CCH, CCH);
    }

    // edge relations: (src features, tgt features)
    const float* xs[4] = { f0, f1, f0, f1 };
    const float* xt[4] = { f0, f1, f1, f2 };
    for (int r = 0; r < 4; ++r) {
        int b = relbase[r];
        int tiles = (E[r] + 15) / 16;
        int blocks = (tiles + 3) / 4;
        edge_mlp_kernel<<<blocks, 128, 0, stream>>>(
            xs[r], xt[r],
            (const int*)d_in[b], (const int*)d_in[b + 1],
            (const float*)d_in[b + 2], ninv[r],
            w1t[r], (const float*)d_in[b + 4],
            w2t[r], (const float*)d_in[b + 6],
            (const float*)d_in[b + 7], (const float*)d_in[b + 8],
            aggp[r], E[r]);
    }

    // updates: h0=[f0,m_a0], h1=[f1,m_a1,m_i1], h2=[f2,m_i2]; out = f + h
    float* out = (float*)d_out;
    int tiles = (N + 15) / 16;
    int ublocks = (tiles + 1) / 2;
    update_kernel<<<ublocks, 64, 0, stream>>>(f0, aggp[0], nullptr, uw1t[0], 256,
        (const float*)d_in[40], uw2t[0], (const float*)d_in[42], out, N);
    update_kernel<<<ublocks, 64, 0, stream>>>(f1, aggp[1], aggp[2], uw1t[1], 384,
        (const float*)d_in[44], uw2t[1], (const float*)d_in[46], out + NC, N);
    update_kernel<<<ublocks, 64, 0, stream>>>(f2, aggp[3], nullptr, uw1t[2], 256,
        (const float*)d_in[48], uw2t[2], (const float*)d_in[50], out + 2 * NC, N);
}
